// sRPEAttention_30537217474912
// MI455X (gfx1250) — matrix operand order repaired and checked by reading, not compiled, not run on hardware
//
#include <hip/hip_runtime.h>
#include <cmath>

#define QSCALE 0.125f      // F^-0.5 = 1/8

typedef __attribute__((ext_vector_type(16))) _Float16 v16h;
typedef __attribute__((ext_vector_type(8)))  _Float16 v8h;
typedef __attribute__((ext_vector_type(8)))  float    v8f;

// ---------------------------------------------------------------- converts
__global__ void cvt_f32_f16(const float* __restrict__ src,
                            _Float16* __restrict__ dst, int n) {
  int i = blockIdx.x * blockDim.x + threadIdx.x;
  if (i < n) dst[i] = (_Float16)src[i];
}

// ------------------------------------------- RPE hidden layer: gelu(feats@W1^T+b1)
// mid[mlp][j][c], j in [0,1023] maps d = j-511 ; row j=1023 zero-padded.
__global__ void rpe_mid_kernel(const float* __restrict__ Wq1, const float* __restrict__ bq1,
                               const float* __restrict__ Wk1, const float* __restrict__ bk1,
                               const float* __restrict__ Wv1, const float* __restrict__ bv1,
                               _Float16* __restrict__ mid) {
  int i = blockIdx.x * blockDim.x + threadIdx.x;   // 3 * 1024 * 512
  int c   = i & 511;
  int j   = (i >> 9) & 1023;
  int mlp = i >> 19;
  if (mlp >= 3) return;
  _Float16 out = (_Float16)0.f;
  if (j < 1023) {
    float d  = (float)(j - 511);
    float f0 = log1pf(fmaxf(d, 0.f));
    float f1 = log1pf(fmaxf(-d, 0.f));
    float f2 = (d == 0.f) ? 1.f : 0.f;
    const float* W1 = (mlp == 0) ? Wq1 : (mlp == 1) ? Wk1 : Wv1;
    const float* b1 = (mlp == 0) ? bq1 : (mlp == 1) ? bk1 : bv1;
    float e = f0 * W1[c * 3 + 0] + f1 * W1[c * 3 + 1] + f2 * W1[c * 3 + 2] + b1[c];
    float u = 0.7978845608028654f * (e + 0.044715f * e * e * e);
    out = (_Float16)(0.5f * e * (1.f + tanhf(u)));
  }
  mid[i] = out;
}

// ---------------------------------------------------------- generic WMMA GEMM
// D[m,n] = outScale * ( sum_k A[m,k] * B(k,n) + bias[n] )
// A row-major f16 (lda).  B(k,n) = Bm[bOff + k + n*bnS]  (K always contiguous).
// 32x32 macro-tile per wave (2x2 of 16x16 WMMA tiles); 4 waves / block.
// Fragment layouts per ISA 7.12.2 (f16 16x16x32).
__global__ __launch_bounds__(128) void wmma_gemm(
    const _Float16* __restrict__ A, int lda, long long aBS,
    const _Float16* __restrict__ Bm, int bnS, long long bBSy, long long bBSz,
    const float* __restrict__ bias,
    float* __restrict__ Cf, _Float16* __restrict__ Ch, int ldc, long long cBS,
    int tilesN, int K, float outScale) {
  const int wave = threadIdx.x >> 5;
  const int lane = threadIdx.x & 31;
  const int lr   = lane & 15;
  const int hi   = lane >> 4;
  const int tile = blockIdx.x * 4 + wave;
  const int tm   = (tile / tilesN) * 32;
  const int tn   = (tile % tilesN) * 32;
  const long long batch = (long long)blockIdx.z * gridDim.y + blockIdx.y;

  const _Float16* A0 = A + batch * aBS + (long long)(tm + lr) * lda;
  const _Float16* A1 = A0 + 16LL * lda;
  const _Float16* B0 = Bm + (long long)blockIdx.y * bBSy + (long long)blockIdx.z * bBSz
                          + (long long)(tn + lr) * bnS;
  const _Float16* B1 = B0 + 16LL * bnS;

  v8f acc00 = {}, acc01 = {}, acc10 = {}, acc11 = {};
  for (int k0 = 0; k0 < K; k0 += 32) {
    // A 16x32 f16 layout: lane(lr,hi) holds K = [hi*8,+8) u [16+hi*8,+8)
    const int ka = k0 + hi * 8;
    // B 32x16 layout: lane lr = column, holds K = [hi*8,+8) u [16+hi*8,+8)
    const int kb = k0 + hi * 8;
    v8h a00 = *(const v8h*)(A0 + ka);
    v8h a01 = *(const v8h*)(A0 + ka + 16);
    v8h a10 = *(const v8h*)(A1 + ka);
    v8h a11 = *(const v8h*)(A1 + ka + 16);
    v8h b00 = *(const v8h*)(B0 + kb);
    v8h b01 = *(const v8h*)(B0 + kb + 16);
    v8h b10 = *(const v8h*)(B1 + kb);
    v8h b11 = *(const v8h*)(B1 + kb + 16);
    v16h af0 = __builtin_shufflevector(a00, a01, 0,1,2,3,4,5,6,7,8,9,10,11,12,13,14,15);
    v16h af1 = __builtin_shufflevector(a10, a11, 0,1,2,3,4,5,6,7,8,9,10,11,12,13,14,15);
    v16h bf0 = __builtin_shufflevector(b00, b01, 0,1,2,3,4,5,6,7,8,9,10,11,12,13,14,15);
    v16h bf1 = __builtin_shufflevector(b10, b11, 0,1,2,3,4,5,6,7,8,9,10,11,12,13,14,15);
    if (k0 + 32 < K) {
      __builtin_prefetch(A0 + ka + 32, 0, 0);   // global_prefetch_b8
      __builtin_prefetch(B0 + kb + 32, 0, 0);
    }
    acc00 = __builtin_amdgcn_wmma_f32_16x16x32_f16(false, af0, false, bf0, (short)0, acc00, false, false);
    acc01 = __builtin_amdgcn_wmma_f32_16x16x32_f16(false, af0, false, bf1, (short)0, acc01, false, false);
    acc10 = __builtin_amdgcn_wmma_f32_16x16x32_f16(false, af1, false, bf0, (short)0, acc10, false, false);
    acc11 = __builtin_amdgcn_wmma_f32_16x16x32_f16(false, af1, false, bf1, (short)0, acc11, false, false);
  }
  // C/D layout: acc[r] = D[r + 8*hi][lr]
  const int n0 = tn + lr, n1 = tn + 16 + lr;
  const float bv0 = bias ? bias[n0] : 0.f;
  const float bv1 = bias ? bias[n1] : 0.f;
  const long long cb = batch * cBS;
#pragma unroll
  for (int r = 0; r < 8; ++r) {
    const int m0 = tm + r + hi * 8;
    const int m1 = m0 + 16;
    const float v00 = (acc00[r] + bv0) * outScale;
    const float v01 = (acc01[r] + bv1) * outScale;
    const float v10 = (acc10[r] + bv0) * outScale;
    const float v11 = (acc11[r] + bv1) * outScale;
    if (Cf) {
      Cf[cb + (long long)m0 * ldc + n0] = v00;
      Cf[cb + (long long)m0 * ldc + n1] = v01;
      Cf[cb + (long long)m1 * ldc + n0] = v10;
      Cf[cb + (long long)m1 * ldc + n1] = v11;
    } else {
      Ch[cb + (long long)m0 * ldc + n0] = (_Float16)v00;
      Ch[cb + (long long)m0 * ldc + n1] = (_Float16)v01;
      Ch[cb + (long long)m1 * ldc + n0] = (_Float16)v10;
      Ch[cb + (long long)m1 * ldc + n1] = (_Float16)v11;
    }
  }
}

// ------------- scatter qkv (B*T,1536) -> q(scaled) (B,H,T,F), k (B,H,T,F),
//               v transposed (B,H,F,T) so the a@v GEMM has contiguous K.
__global__ void qkv_remap(const _Float16* __restrict__ qkvh,
                          _Float16* __restrict__ qh, _Float16* __restrict__ kh,
                          _Float16* __restrict__ vh) {
  int i = blockIdx.x * blockDim.x + threadIdx.x;  // 512*1536
  int nn = i % 1536, row = i / 1536;
  int which = nn / 512, rem = nn % 512;
  int h = rem / 64, f = rem % 64;
  int b = row / 256, t = row % 256;
  float v = (float)qkvh[i];
  long long bh = b * 8 + h;
  if (which == 0)      qh[(bh * 256 + t) * 64 + f] = (_Float16)(v * QSCALE);
  else if (which == 1) kh[(bh * 256 + t) * 64 + f] = (_Float16)v;
  else                 vh[(bh * 64 + f) * 256 + t] = (_Float16)v;
}

// -------------------- fused: attn + gathered Pk/Pq + mask + softmax -> a (f16)
// one wave32 per (b,h,t) row of 256 scores
__global__ __launch_bounds__(256) void softmax_kernel(
    const float* __restrict__ attn, const _Float16* __restrict__ Pk,
    const _Float16* __restrict__ Pq, const float* __restrict__ mask,
    const int* __restrict__ frame, _Float16* __restrict__ ah) {
  const int wave = threadIdx.x >> 5;
  const int lane = threadIdx.x & 31;
  const int row  = blockIdx.x * 8 + wave;   // bh*256 + t
  const int t  = row & 255;
  const int bh = row >> 8;
  const int b  = bh >> 3;
  const int fi_t = frame[b * 256 + t];
  float w[8];
#pragma unroll
  for (int i = 0; i < 8; ++i) {
    const int s = i * 32 + lane;
    const int j = fi_t - frame[b * 256 + s] + 511;   // in [0,1022]
    float x = attn[(long long)row * 256 + s]
            + (float)Pk[(long long)row * 1024 + j]
            + (float)Pq[((long long)bh * 256 + s) * 1024 + j];
    if (mask[(long long)b * 65536 + t * 256 + s] == 0.f) x = -__builtin_inff();
    w[i] = x;
  }
  float m = w[0];
#pragma unroll
  for (int i = 1; i < 8; ++i) m = fmaxf(m, w[i]);
  for (int off = 16; off >= 1; off >>= 1) m = fmaxf(m, __shfl_xor(m, off, 32));
  float e[8], sum;
  if (m == -__builtin_inff()) {           // fully-masked row -> uniform (ref semantics)
#pragma unroll
    for (int i = 0; i < 8; ++i) e[i] = 1.f;
    sum = 256.f;
  } else {
    sum = 0.f;
#pragma unroll
    for (int i = 0; i < 8; ++i) { e[i] = __expf(w[i] - m); sum += e[i]; }
    for (int off = 16; off >= 1; off >>= 1) sum += __shfl_xor(sum, off, 32);
  }
  const float inv = 1.f / sum;
#pragma unroll
  for (int i = 0; i < 8; ++i)
    ah[(long long)row * 256 + i * 32 + lane] = (_Float16)(e[i] * inv);
}

// ------------- out += sum_s a[t,s] * Rv_tab[d(t,s), h*64+f]; pack pre-proj (f16)
// a-row staged into LDS via CDNA5 async copy (ASYNCcnt-tracked).
__global__ __launch_bounds__(64) void rpe_v_kernel(
    const _Float16* __restrict__ ah, const float* __restrict__ oattn,
    const _Float16* __restrict__ rtab_v, const int* __restrict__ frame,
    _Float16* __restrict__ preh) {
  __shared__ __align__(16) _Float16 sah[256];
  __shared__ int sj[256];
  const int bht = blockIdx.x;               // (b*8+h)*256+t
  const int t = bht & 255;
  const int h = (bht >> 8) & 7;
  const int b = bht >> 11;
  // wave 0: async-copy the 512B probability row into LDS (16B per lane)
  if (threadIdx.x < 32) {
    const _Float16* gsrc = ah + (long long)bht * 256 + threadIdx.x * 8;
    unsigned lds_off = (unsigned)(size_t)(&sah[threadIdx.x * 8]);
    asm volatile("global_load_async_to_lds_b128 %0, %1, off"
                 :: "v"(lds_off), "v"(gsrc) : "memory");
  }
  const int fi_t = frame[b * 256 + t];
  for (int s = threadIdx.x; s < 256; s += 64)
    sj[s] = (fi_t - frame[b * 256 + s] + 511) * 512;
  asm volatile("s_wait_asynccnt 0" ::: "memory");
  __syncthreads();
  const int f = threadIdx.x;
  const _Float16* Rv = rtab_v + h * 64 + f;
  float acc = 0.f;
  for (int s = 0; s < 256; ++s) acc += (float)sah[s] * (float)Rv[sj[s]];
  preh[((long long)(b * 256 + t)) * 512 + h * 64 + f] =
      (_Float16)(oattn[(long long)bht * 64 + f] + acc);
}

// ================================================================ launcher
extern "C" void kernel_launch(void* const* d_in, const int* in_sizes, int n_in,
                              void* d_out, int out_size, void* d_ws, size_t ws_size,
                              hipStream_t stream) {
  (void)in_sizes; (void)n_in; (void)out_size; (void)ws_size;
  const float* x     = (const float*)d_in[0];
  const float* mask  = (const float*)d_in[1];
  const int*   frame = (const int*)  d_in[2];
  const float* Wqkv  = (const float*)d_in[3];
  const float* bqkv  = (const float*)d_in[4];
  const float* Wproj = (const float*)d_in[5];
  const float* bproj = (const float*)d_in[6];
  const float* Wq1 = (const float*)d_in[7];
  const float* bq1 = (const float*)d_in[8];
  const float* Wq2 = (const float*)d_in[9];
  const float* bq2 = (const float*)d_in[10];
  const float* Wk1 = (const float*)d_in[11];
  const float* bk1 = (const float*)d_in[12];
  const float* Wk2 = (const float*)d_in[13];
  const float* bk2 = (const float*)d_in[14];
  const float* Wv1 = (const float*)d_in[15];
  const float* bv1 = (const float*)d_in[16];
  const float* Wv2 = (const float*)d_in[17];
  const float* bv2 = (const float*)d_in[18];

  char* p = (char*)d_ws;
  auto alloc = [&](size_t bytes) { void* r = (void*)p; p += (bytes + 255) & ~(size_t)255; return r; };
  _Float16* xh     = (_Float16*)alloc(512 * 512 * 2);
  _Float16* wqkvh  = (_Float16*)alloc(1536 * 512 * 2);
  _Float16* wprojh = (_Float16*)alloc(512 * 512 * 2);
  _Float16* w2h    = (_Float16*)alloc(3 * 512 * 512 * 2);      // q,k,v
  _Float16* midh   = (_Float16*)alloc(3 * 1024 * 512 * 2);
  _Float16* rtabh  = (_Float16*)alloc(3 * 1024 * 512 * 2);     // q,k,v tables
  _Float16* qkvh   = (_Float16*)alloc(512 * 1536 * 2);
  _Float16* qh     = (_Float16*)alloc(262144 * 2);
  _Float16* kh     = (_Float16*)alloc(262144 * 2);
  _Float16* vh     = (_Float16*)alloc(262144 * 2);             // (B,H,F,T)
  _Float16* ph_k   = (_Float16*)alloc((size_t)16 * 256 * 1024 * 2);
  _Float16* ph_q   = (_Float16*)alloc((size_t)16 * 256 * 1024 * 2);
  _Float16* ahb    = (_Float16*)alloc((size_t)16 * 256 * 256 * 2);
  _Float16* preh   = (_Float16*)alloc(512 * 512 * 2);
  float*    attnf  = (float*)alloc((size_t)16 * 256 * 256 * 4);
  float*    oattnf = (float*)alloc((size_t)16 * 256 * 64 * 4);

  // 1) f32 -> f16 operand conversion
  cvt_f32_f16<<<1024, 256, 0, stream>>>(x, xh, 262144);
  cvt_f32_f16<<<3072, 256, 0, stream>>>(Wqkv, wqkvh, 786432);
  cvt_f32_f16<<<1024, 256, 0, stream>>>(Wproj, wprojh, 262144);
  cvt_f32_f16<<<1024, 256, 0, stream>>>(Wq2, w2h + 0 * 262144, 262144);
  cvt_f32_f16<<<1024, 256, 0, stream>>>(Wk2, w2h + 1 * 262144, 262144);
  cvt_f32_f16<<<1024, 256, 0, stream>>>(Wv2, w2h + 2 * 262144, 262144);

  // 2) RPE hidden layer over the 1023 distinct d values
  rpe_mid_kernel<<<6144, 256, 0, stream>>>(Wq1, bq1, Wk1, bk1, Wv1, bv1, midh);

  // 3) R tables: rtab[mlp] = mid[mlp] @ W2[mlp]^T + b2   (M=1024,N=512,K=512)
  //    k-scale of the Rq score term folded into the q-table (outScale).
  const float* b2s[3] = {bq2, bk2, bv2};
  for (int mlp = 0; mlp < 3; ++mlp)
    wmma_gemm<<<dim3(128, 1, 1), 128, 0, stream>>>(
        midh + mlp * 524288, 512, 0LL,
        w2h + mlp * 262144, 512, 0LL, 0LL,
        b2s[mlp], nullptr, rtabh + mlp * 524288, 512, 0LL,
        16, 512, (mlp == 0) ? QSCALE : 1.f);

  // 4) qkv = x @ Wqkv^T + bqkv   (M=512,N=1536,K=512)
  wmma_gemm<<<dim3(192, 1, 1), 128, 0, stream>>>(
      xh, 512, 0LL, wqkvh, 512, 0LL, 0LL, bqkv,
      nullptr, qkvh, 1536, 0LL, 48, 512, 1.f);
  qkv_remap<<<3072, 256, 0, stream>>>(qkvh, qh, kh, vh);

  // 5) attn = q_scaled @ k^T per (b,h)   (M=256,N=256,K=64), 16 batches
  wmma_gemm<<<dim3(16, 16, 1), 128, 0, stream>>>(
      qh, 64, 16384LL, kh, 64, 16384LL, 0LL, nullptr,
      attnf, nullptr, 256, 65536LL, 8, 64, 1.f);

  // 6) Pk = q_scaled @ Rk_tab_h^T ; Pq = k @ (scaled Rq_tab_h)^T  (M=256,N=1024,K=64)
  wmma_gemm<<<dim3(64, 8, 2), 128, 0, stream>>>(
      qh, 64, 16384LL, rtabh + 1 * 524288, 512, 64LL, 0LL, nullptr,
      nullptr, ph_k, 1024, 262144LL, 32, 64, 1.f);
  wmma_gemm<<<dim3(64, 8, 2), 128, 0, stream>>>(
      kh, 64, 16384LL, rtabh + 0 * 524288, 512, 64LL, 0LL, nullptr,
      nullptr, ph_q, 1024, 262144LL, 32, 64, 1.f);

  // 7) fused gather + masked softmax
  softmax_kernel<<<512, 256, 0, stream>>>(attnf, ph_k, ph_q, mask, frame, ahb);

  // 8) out_attn = a @ v per (b,h)  (M=256,N=64,K=256), v pre-transposed
  wmma_gemm<<<dim3(4, 16, 1), 128, 0, stream>>>(
      ahb, 256, 65536LL, vh, 256, 16384LL, 0LL, nullptr,
      oattnf, nullptr, 64, 16384LL, 2, 256, 1.f);

  // 9) + a-weighted gathered Rv rows; pack pre-projection activations
  rpe_v_kernel<<<4096, 64, 0, stream>>>(ahb, oattnf, rtabh + 2 * 524288, frame, preh);

  // 10) final: out = pre @ Wproj^T + bproj  (M=512,N=512,K=512) -> f32 d_out
  wmma_gemm<<<dim3(64, 1, 1), 128, 0, stream>>>(
      preh, 512, 0LL, wprojh, 512, 0LL, 0LL, bproj,
      (float*)d_out, nullptr, 512, 0LL, 16, 512, 1.f);
}
